// ProbSparseAttention_28097676050516
// MI455X (gfx1250) — matrix operand order repaired and checked by reading, not compiled, not run on hardware
//
#include <hip/hip_runtime.h>
#include <hip/hip_bf16.h>

typedef __attribute__((ext_vector_type(16))) _Float16 v16h;
typedef __attribute__((ext_vector_type(8)))  _Float16 v8h;
typedef __attribute__((ext_vector_type(8)))  float    v8f;
typedef __attribute__((ext_vector_type(4)))  int      i32x4;

#define BB 2
#define TT 2048
#define CC 1024
#define HH 16
#define DD 64
#define ATT_SCALE 0.125f   // 1/sqrt(64)

#if __has_builtin(__builtin_amdgcn_global_load_async_to_lds_b128) && \
    __has_builtin(__builtin_amdgcn_s_wait_asynccnt)
#define ASYNC_OK 1
#endif

// ---------------------------------------------------------------------------
// async global -> LDS copy of 16 bytes (per lane), CDNA5 path when available
// ---------------------------------------------------------------------------
__device__ __forceinline__ void g2l16(_Float16* lds, const _Float16* g) {
#ifdef ASYNC_OK
  __builtin_amdgcn_global_load_async_to_lds_b128(
      (__attribute__((address_space(1))) i32x4*)(void*)g,
      (__attribute__((address_space(3))) i32x4*)(void*)lds, 0, 0);
#else
  *(v8h*)lds = *(const v8h*)g;
#endif
}
__device__ __forceinline__ void wait_async_all() {
#ifdef ASYNC_OK
  __builtin_amdgcn_s_wait_asynccnt(0);
#endif
}

// ---------------------------------------------------------------------------
// WMMA helpers
// ---------------------------------------------------------------------------
__device__ __forceinline__ v8f wmma_f16(v16h a, v16h b, v8f c) {
  return __builtin_amdgcn_wmma_f32_16x16x32_f16(false, a, false, b,
                                                (short)0, c, false, false);
}

// A fragment: 16(M) x 32(K) f16. Lane l holds row M=l&15.
// lanes 0-15 : halves 0-7 -> K=0..7,  halves 8-15 -> K=16..23
// lanes 16-31: halves 0-7 -> K=8..15, halves 8-15 -> K=24..31
__device__ __forceinline__ v16h ldfragA(const _Float16* p0, int ld, int lane) {
  const _Float16* p = p0 + (size_t)(lane & 15) * ld + ((lane < 16) ? 0 : 8);
  v8h lo = *(const v8h*)p;
  v8h hi = *(const v8h*)(p + 16);
  v16h r;
#pragma unroll
  for (int i = 0; i < 8; ++i) { r[i] = lo[i]; r[i + 8] = hi[i]; }
  return r;
}

// B fragment: 32(K) x 16(N) f16, sourced from [N][K]-major storage.
// Lane l holds col N=l&15; lanes 0-15 hold K=0..7,16..23, lanes 16-31 hold K=8..15,24..31.
__device__ __forceinline__ v16h ldfragB(const _Float16* p0, int ld, int lane) {
  const _Float16* p = p0 + (size_t)(lane & 15) * ld + ((lane < 16) ? 0 : 8);
  v8h lo = *(const v8h*)p;
  v8h hi = *(const v8h*)(p + 16);
  v16h r;
#pragma unroll
  for (int i = 0; i < 8; ++i) { r[i] = lo[i]; r[i + 8] = hi[i]; }
  return r;
}

// ---------------------------------------------------------------------------
// fp32 -> f16 conversion (x)
// ---------------------------------------------------------------------------
__global__ void cvt_f32_f16(const float* __restrict__ in,
                            _Float16* __restrict__ out, int n) {
  int i = blockIdx.x * 256 + threadIdx.x;
  if (i < n) out[i] = (_Float16)in[i];
}

// fp32 [K][N] -> f16 transposed [N][K] (weights, done once)
__global__ __launch_bounds__(256) void cvt_w_t(const float* __restrict__ in,
                                               _Float16* __restrict__ out) {
  __shared__ float tile[32][33];
  const int n0 = blockIdx.x * 32, k0 = blockIdx.y * 32;
  const int tx = threadIdx.x & 31, ty = threadIdx.x >> 5;  // 32 x 8
#pragma unroll
  for (int r = ty; r < 32; r += 8)
    tile[r][tx] = in[(size_t)(k0 + r) * CC + n0 + tx];
  __syncthreads();
#pragma unroll
  for (int r = ty; r < 32; r += 8)
    out[(size_t)(n0 + r) * CC + k0 + tx] = (_Float16)tile[tx][r];
}

// ---------------------------------------------------------------------------
// QKV projection GEMM:  Y[4096,1024] = X16 @ W^T16' + b   (Wt is [N][K] f16)
// blockIdx.z: 0 -> Q ([B,H,T,D]), 1 -> K ([B,H,T,D]), 2 -> V^T ([B,H,D,T])
// WG tile 128x128, 8 waves, wave tile 32x64, K-step 32, double-buffered LDS.
// ---------------------------------------------------------------------------
__global__ __launch_bounds__(256) void qkv_gemm(
    const _Float16* __restrict__ X,
    const _Float16* __restrict__ Wqt, const _Float16* __restrict__ Wkt,
    const _Float16* __restrict__ Wvt,
    const float* __restrict__ bq, const float* __restrict__ bk,
    const float* __restrict__ bv,
    _Float16* __restrict__ Qo, _Float16* __restrict__ Ko,
    _Float16* __restrict__ Vt) {
  __shared__ __align__(16) _Float16 lx[2][128 * 40];  // X tile   [m][k]
  __shared__ __align__(16) _Float16 lw[2][128 * 40];  // W^T tile [n][k]

  const int tid = threadIdx.x, lane = tid & 31, wid = tid >> 5;
  const int wm = wid & 3, wn = wid >> 2;          // 4 waves on M, 2 on N
  const int n0 = blockIdx.x * 128, m0 = blockIdx.y * 128;
  const int z = blockIdx.z;
  const _Float16* Wt = (z == 0) ? Wqt : (z == 1) ? Wkt : Wvt;

  // each thread copies 2 fixed b128 chunks per tile (128 rows x 32 halves)
  const int r0 = tid >> 2, c0 = (tid & 3) * 8;            // chunk 0
  const int r1 = (tid + 256) >> 2, c1 = (tid & 3) * 8;    // chunk 1

  v8f acc[2][4] = {};

  // prologue: fill buffer 0
  g2l16(&lx[0][r0 * 40 + c0], &X[(size_t)(m0 + r0) * CC + c0]);
  g2l16(&lx[0][r1 * 40 + c1], &X[(size_t)(m0 + r1) * CC + c1]);
  g2l16(&lw[0][r0 * 40 + c0], &Wt[(size_t)(n0 + r0) * CC + c0]);
  g2l16(&lw[0][r1 * 40 + c1], &Wt[(size_t)(n0 + r1) * CC + c1]);
  wait_async_all();
  __syncthreads();

  int pp = 0;
  for (int k0 = 0; k0 < CC; k0 += 32) {
    if (k0 + 32 < CC) {  // issue next tile into the other buffer
      int kn = k0 + 32, q = pp ^ 1;
      g2l16(&lx[q][r0 * 40 + c0], &X[(size_t)(m0 + r0) * CC + kn + c0]);
      g2l16(&lx[q][r1 * 40 + c1], &X[(size_t)(m0 + r1) * CC + kn + c1]);
      g2l16(&lw[q][r0 * 40 + c0], &Wt[(size_t)(n0 + r0) * CC + kn + c0]);
      g2l16(&lw[q][r1 * 40 + c1], &Wt[(size_t)(n0 + r1) * CC + kn + c1]);
    }
    v16h a[2], b[4];
#pragma unroll
    for (int i = 0; i < 2; ++i)
      a[i] = ldfragA(&lx[pp][(wm * 32 + i * 16) * 40], 40, lane);
#pragma unroll
    for (int j = 0; j < 4; ++j)
      b[j] = ldfragB(&lw[pp][(wn * 64 + j * 16) * 40], 40, lane);
#pragma unroll
    for (int i = 0; i < 2; ++i)
#pragma unroll
      for (int j = 0; j < 4; ++j) acc[i][j] = wmma_f16(a[i], b[j], acc[i][j]);
    wait_async_all();
    __syncthreads();
    pp ^= 1;
  }

  const float* bias = (z == 0) ? bq : (z == 1) ? bk : bv;
  _Float16* out = (z == 0) ? Qo : (z == 1) ? Ko : Vt;

#pragma unroll
  for (int i = 0; i < 2; ++i) {
#pragma unroll
    for (int j = 0; j < 4; ++j) {
      int ncol = n0 + wn * 64 + j * 16 + (lane & 15);
      float bv_ = bias[ncol];
      int h = ncol >> 6, d = ncol & 63;
#pragma unroll
      for (int f = 0; f < 8; ++f) {
        int m = m0 + wm * 32 + i * 16 + ((lane < 16) ? f : f + 8);
        int bidx = m >> 11, t = m & (TT - 1);
        float val = acc[i][j][f] + bv_;
        size_t addr;
        if (z == 2)  // V stored transposed: [B,H,D,T]
          addr = ((size_t)(bidx * HH + h) * DD + d) * TT + t;
        else         // Q,K: [B,H,T,D]
          addr = ((size_t)(bidx * HH + h) * TT + t) * DD + d;
        out[addr] = (_Float16)val;
      }
    }
  }
}

// ---------------------------------------------------------------------------
// Flash attention per (b,h). WG = 128 queries (16 per wave), KB = 32 keys/iter,
// double-buffered K/V tiles with async copies.
// Q:[B,H,T,D] f16, K:[B,H,T,D] f16, Vt:[B,H,D,T] f16 -> O:[B,T,C] f16
// ---------------------------------------------------------------------------
__global__ __launch_bounds__(256) void flash_attn(
    const _Float16* __restrict__ Q, const _Float16* __restrict__ K,
    const _Float16* __restrict__ Vt, _Float16* __restrict__ O) {
  __shared__ __align__(16) _Float16 lk[2][32 * 72];     // K tile  [key][d]
  __shared__ __align__(16) _Float16 lv[2][64 * 40];     // V^T tile [d][key]
  __shared__ __align__(16) _Float16 lp[8 * 16 * 40];    // P staging per wave

  const int tid = threadIdx.x, lane = tid & 31, wid = tid >> 5;
  const int bh = blockIdx.y;                            // b*16 + h
  const int q0 = blockIdx.x * 128;

  const _Float16* Qh = Q + ((size_t)bh * TT + q0 + wid * 16) * DD;
  const _Float16* Kh = K + (size_t)bh * TT * DD;
  const _Float16* Vh = Vt + (size_t)bh * DD * TT;

  v16h qf[2];
  qf[0] = ldfragA(Qh, DD, lane);        // d = 0..31
  qf[1] = ldfragA(Qh + 32, DD, lane);   // d = 32..63

  v8f o[4] = {};
  float m[8], l[8];
#pragma unroll
  for (int f = 0; f < 8; ++f) { m[f] = -1e30f; l[f] = 0.0f; }

  _Float16* lpw = &lp[wid * 16 * 40];

  // per-thread coop copy positions: one b128 per tile per thread
  const int kr = tid >> 3, kc = (tid & 7) * 8;   // K tile: 32 x 64
  const int vd = tid >> 2, vc = (tid & 3) * 8;   // V^T tile: 64 x 32

  // prologue: key block 0 into buffer 0
  g2l16(&lk[0][kr * 72 + kc], &Kh[(size_t)kr * DD + kc]);
  g2l16(&lv[0][vd * 40 + vc], &Vh[(size_t)vd * TT + vc]);
  wait_async_all();
  __syncthreads();

  int pp = 0;
  for (int t0 = 0; t0 < TT; t0 += 32) {
    if (t0 + 32 < TT) {  // issue next key block
      int tn = t0 + 32, qb = pp ^ 1;
      g2l16(&lk[qb][kr * 72 + kc], &Kh[(size_t)(tn + kr) * DD + kc]);
      g2l16(&lv[qb][vd * 40 + vc], &Vh[(size_t)vd * TT + tn + vc]);
    }

    // S = Q @ K^T over this key block (two 16-key tiles, K-dim = 64)
    v8f s[2];
#pragma unroll
    for (int kt = 0; kt < 2; ++kt) {
      v8f sa = {};
      sa = wmma_f16(qf[0], ldfragB(&lk[pp][kt * 16 * 72 + 0], 72, lane), sa);
      sa = wmma_f16(qf[1], ldfragB(&lk[pp][kt * 16 * 72 + 32], 72, lane), sa);
      s[kt] = sa;
    }

    // online softmax
    float p0[8], p1[8];
#pragma unroll
    for (int f = 0; f < 8; ++f) {
      float a0 = s[0][f] * ATT_SCALE;
      float a1 = s[1][f] * ATT_SCALE;
      float rm = fmaxf(a0, a1);
      rm = fmaxf(rm, __shfl_xor(rm, 1));
      rm = fmaxf(rm, __shfl_xor(rm, 2));
      rm = fmaxf(rm, __shfl_xor(rm, 4));
      rm = fmaxf(rm, __shfl_xor(rm, 8));   // reduce across 16-lane half
      float mn = fmaxf(m[f], rm);
      float al = __expf(m[f] - mn);
      m[f] = mn;
      p0[f] = __expf(a0 - mn);
      p1[f] = __expf(a1 - mn);
      float rs = p0[f] + p1[f];
      rs += __shfl_xor(rs, 1);
      rs += __shfl_xor(rs, 2);
      rs += __shfl_xor(rs, 4);
      rs += __shfl_xor(rs, 8);
      l[f] = l[f] * al + rs;
#pragma unroll
      for (int dt = 0; dt < 4; ++dt) o[dt][f] *= al;
    }

    // stage P (acc layout) -> [q][key] LDS so it can re-enter as an A fragment
#pragma unroll
    for (int f = 0; f < 8; ++f) {
      int r = (lane < 16) ? f : f + 8;
      int c = lane & 15;
      lpw[r * 40 + c] = (_Float16)p0[f];
      lpw[r * 40 + 16 + c] = (_Float16)p1[f];
    }
    v16h pf = ldfragA(lpw, 40, lane);    // DS ops are in-order within a wave

    // O += P @ V  (K-dim = 32 keys, 4 tiles over d)
#pragma unroll
    for (int dt = 0; dt < 4; ++dt) {
      v16h vf = ldfragB(&lv[pp][dt * 16 * 40], 40, lane);
      o[dt] = wmma_f16(pf, vf, o[dt]);
    }
    wait_async_all();
    __syncthreads();
    pp ^= 1;
  }

  // epilogue: normalize and scatter to [B,T,C] f16
  int b = bh >> 4, h = bh & 15;
#pragma unroll
  for (int dt = 0; dt < 4; ++dt) {
    int col = h * DD + dt * 16 + (lane & 15);
#pragma unroll
    for (int f = 0; f < 8; ++f) {
      int q = q0 + wid * 16 + ((lane < 16) ? f : f + 8);
      O[(size_t)(b * TT + q) * CC + col] = (_Float16)(o[dt][f] / l[f]);
    }
  }
}

// ---------------------------------------------------------------------------
// Output projection: out_f32[4096,1024] = A16 @ Wo^T16' + bo  (Wt is [N][K])
// ---------------------------------------------------------------------------
__global__ __launch_bounds__(256) void out_gemm(
    const _Float16* __restrict__ X, const _Float16* __restrict__ Wt,
    const float* __restrict__ bias, float* __restrict__ out) {
  __shared__ __align__(16) _Float16 lx[2][128 * 40];
  __shared__ __align__(16) _Float16 lw[2][128 * 40];

  const int tid = threadIdx.x, lane = tid & 31, wid = tid >> 5;
  const int wm = wid & 3, wn = wid >> 2;
  const int n0 = blockIdx.x * 128, m0 = blockIdx.y * 128;

  const int r0 = tid >> 2, c0 = (tid & 3) * 8;
  const int r1 = (tid + 256) >> 2, c1 = (tid & 3) * 8;

  v8f acc[2][4] = {};

  g2l16(&lx[0][r0 * 40 + c0], &X[(size_t)(m0 + r0) * CC + c0]);
  g2l16(&lx[0][r1 * 40 + c1], &X[(size_t)(m0 + r1) * CC + c1]);
  g2l16(&lw[0][r0 * 40 + c0], &Wt[(size_t)(n0 + r0) * CC + c0]);
  g2l16(&lw[0][r1 * 40 + c1], &Wt[(size_t)(n0 + r1) * CC + c1]);
  wait_async_all();
  __syncthreads();

  int pp = 0;
  for (int k0 = 0; k0 < CC; k0 += 32) {
    if (k0 + 32 < CC) {
      int kn = k0 + 32, q = pp ^ 1;
      g2l16(&lx[q][r0 * 40 + c0], &X[(size_t)(m0 + r0) * CC + kn + c0]);
      g2l16(&lx[q][r1 * 40 + c1], &X[(size_t)(m0 + r1) * CC + kn + c1]);
      g2l16(&lw[q][r0 * 40 + c0], &Wt[(size_t)(n0 + r0) * CC + kn + c0]);
      g2l16(&lw[q][r1 * 40 + c1], &Wt[(size_t)(n0 + r1) * CC + kn + c1]);
    }
    v16h a[2], b[4];
#pragma unroll
    for (int i = 0; i < 2; ++i)
      a[i] = ldfragA(&lx[pp][(wm * 32 + i * 16) * 40], 40, lane);
#pragma unroll
    for (int j = 0; j < 4; ++j)
      b[j] = ldfragB(&lw[pp][(wn * 64 + j * 16) * 40], 40, lane);
#pragma unroll
    for (int i = 0; i < 2; ++i)
#pragma unroll
      for (int j = 0; j < 4; ++j) acc[i][j] = wmma_f16(a[i], b[j], acc[i][j]);
    wait_async_all();
    __syncthreads();
    pp ^= 1;
  }

#pragma unroll
  for (int i = 0; i < 2; ++i) {
#pragma unroll
    for (int j = 0; j < 4; ++j) {
      int ncol = n0 + wn * 64 + j * 16 + (lane & 15);
      float bv_ = bias[ncol];
#pragma unroll
      for (int f = 0; f < 8; ++f) {
        int mrow = m0 + wm * 32 + i * 16 + ((lane < 16) ? f : f + 8);
        out[(size_t)mrow * CC + ncol] = acc[i][j][f] + bv_;
      }
    }
  }
}

// ---------------------------------------------------------------------------
extern "C" void kernel_launch(void* const* d_in, const int* in_sizes, int n_in,
                              void* d_out, int out_size, void* d_ws,
                              size_t ws_size, hipStream_t stream) {
  const float* x  = (const float*)d_in[0];
  const float* Wq = (const float*)d_in[1];
  const float* bq = (const float*)d_in[2];
  const float* Wk = (const float*)d_in[3];
  const float* bk = (const float*)d_in[4];
  const float* Wv = (const float*)d_in[5];
  const float* bv = (const float*)d_in[6];
  const float* Wo = (const float*)d_in[7];
  const float* bo = (const float*)d_in[8];
  float* out = (float*)d_out;

  const size_t NX = (size_t)BB * TT * CC;   // 4,194,304
  const size_t NW = (size_t)CC * CC;        // 1,048,576

  char* ws = (char*)d_ws;
  _Float16* x16  = (_Float16*)ws;
  _Float16* wqT  = x16 + NX;    // [N][K] f16
  _Float16* wkT  = wqT + NW;
  _Float16* wvT  = wkT + NW;
  _Float16* woT  = wvT + NW;
  _Float16* q16  = woT + NW;    // [B,H,T,D]
  _Float16* k16  = q16 + NX;    // [B,H,T,D]
  _Float16* vT16 = k16 + NX;    // [B,H,D,T]
  _Float16* a16  = vT16 + NX;   // attention output [B,T,C]

  cvt_f32_f16<<<(int)((NX + 255) / 256), 256, 0, stream>>>(x, x16, (int)NX);
  cvt_w_t<<<dim3(32, 32), 256, 0, stream>>>(Wq, wqT);
  cvt_w_t<<<dim3(32, 32), 256, 0, stream>>>(Wk, wkT);
  cvt_w_t<<<dim3(32, 32), 256, 0, stream>>>(Wv, wvT);
  cvt_w_t<<<dim3(32, 32), 256, 0, stream>>>(Wo, woT);

  // Q/K/V projections: N-blocks x M-blocks x {Q,K,V}
  qkv_gemm<<<dim3(8, 32, 3), 256, 0, stream>>>(x16, wqT, wkT, wvT, bq, bk, bv,
                                               q16, k16, vT16);

  // attention: 16 query-blocks x 32 (b,h) pairs
  flash_attn<<<dim3(16, 32), 256, 0, stream>>>(q16, k16, vT16, a16);

  // output projection (fp32 out + bias)
  out_gemm<<<dim3(8, 32), 256, 0, stream>>>(a16, woT, bo, out);
}
